// DFUB_31765578121353
// MI455X (gfx1250) — matrix operand order repaired and checked by reading, not compiled, not run on hardware
//
#include <hip/hip_runtime.h>

typedef __bf16 bf16_t;
typedef __attribute__((ext_vector_type(16))) __bf16 v16bf;
typedef __attribute__((ext_vector_type(8)))  __bf16 v8bf;
typedef __attribute__((ext_vector_type(8)))  float  v8f;
typedef __attribute__((ext_vector_type(4)))  unsigned int u32x4;
typedef __attribute__((ext_vector_type(8)))  int i32x8;
typedef __attribute__((ext_vector_type(4)))  int i32x4;

#if defined(__has_builtin)
#if __has_builtin(__builtin_amdgcn_tensor_load_to_lds)
#define CDNA5_HAS_TDM 1
#endif
#endif
#if __has_include(<hip/amd_detail/amd_gfx1250_TDM.h>)
#define CDNA5_TDM_6ARG 1
#endif

__device__ __forceinline__ v8f wmma_bf16f32(v16bf a, v16bf b, v8f c) {
  // 8 args: (neg_a, A, neg_b, B, c_mod, C, reuse_a, reuse_b)
  return __builtin_amdgcn_wmma_f32_16x16x32_bf16(false, a, false, b, (short)0, c, false, false);
}

// A-fragment: 16 bf16/lane as two aligned 8-elem runs (K = khalf+0..7, 16+khalf+0..7)
__device__ __forceinline__ v16bf load_fragA(const bf16_t* lo, const bf16_t* hi) {
  v8bf a = *(const v8bf*)lo;
  v8bf b = *(const v8bf*)hi;
  v16bf r;
#pragma unroll
  for (int i = 0; i < 8; ++i) { r[i] = a[i]; r[i + 8] = b[i]; }
  return r;
}

// B-fragment: p = base + (lane>>4)*16; two 8-elem runs (K = khalf+0..7, 16+khalf+0..7)
__device__ __forceinline__ v16bf load_fragB(const bf16_t* p) {
  const int kh = (int)(threadIdx.x & 16) >> 1;
  v8bf a = *(const v8bf*)(p - kh);
  v8bf b = *(const v8bf*)(p + 16 - kh);
  v16bf r;
#pragma unroll
  for (int i = 0; i < 8; ++i) { r[i] = a[i]; r[i + 8] = b[i]; }
  return r;
}

#ifdef CDNA5_HAS_TDM
// 2D TDM load: tile_h rows x tile_w bf16 elements, global row stride = stride_elems.
// Descriptor layout per CDNA5 ISA 8.3/8.4 (D# group0/group1), data_size=1 (2 bytes).
__device__ __forceinline__ void tdm_load_2d_bf16(unsigned lds_byte_addr, const bf16_t* gptr,
                                                 unsigned tile_w, unsigned tile_h,
                                                 unsigned stride_elems) {
  unsigned long long ga = (unsigned long long)(uintptr_t)gptr;
  u32x4 g0 = {};
  g0[0] = 1u;                                         // count=1, user mode, no gather
  g0[1] = lds_byte_addr;                              // lds_addr [63:32]
  g0[2] = (unsigned)(ga & 0xffffffffu);               // global_addr [95:64]
  g0[3] = (unsigned)((ga >> 32) & 0x01ffffffu) | (2u << 30);  // addr[56:32], type=2
  i32x8 g1 = {};
  g1[0] = (int)(1u << 16);                            // data_size=1 (2B), wg_mask=0
  g1[1] = (int)((tile_w & 0xffffu) << 16);            // tensor_dim0[15:0] @ bit48
  g1[2] = (int)(((tile_w >> 16) & 0xffffu) | ((tile_h & 0xffffu) << 16));  // td0 hi | td1 lo
  g1[3] = (int)(((tile_h >> 16) & 0xffffu) | ((tile_w & 0xffffu) << 16));  // td1 hi | tile_dim0
  g1[4] = (int)(tile_h & 0xffffu);                    // tile_dim1 (tile_dim2=0)
  g1[5] = (int)stride_elems;                          // tensor_dim0_stride[31:0]
  g1[6] = 0;
  g1[7] = 0;
  i32x4 gz = {};
#ifdef CDNA5_TDM_6ARG
  i32x8 gz8 = {};
  __builtin_amdgcn_tensor_load_to_lds(g0, g1, gz, gz, gz8, 0);
#else
  __builtin_amdgcn_tensor_load_to_lds(g0, g1, gz, gz, 0);
#endif
}
#endif

// ---------------- fp32 -> bf16 conversion ----------------
__global__ void cvt_bf16(const float* __restrict__ src, bf16_t* __restrict__ dst, int n) {
  int i = (blockIdx.x * 256 + threadIdx.x) * 4;
  if (i + 3 < n) {
    float4 v = *(const float4*)(src + i);
    dst[i]     = (bf16_t)v.x; dst[i + 1] = (bf16_t)v.y;
    dst[i + 2] = (bf16_t)v.z; dst[i + 3] = (bf16_t)v.w;
  } else {
    for (int j = i; j < n; ++j) dst[j] = (bf16_t)src[j];
  }
}

// ---------------- gate: e_td = concat(E_t,E_d) @ W_td  (fp32, tiny) ----------------
__global__ __launch_bounds__(256) void gate_kernel(
    const float* __restrict__ Et, const float* __restrict__ Ed,
    const float* __restrict__ Wtd, float* __restrict__ gate) {
  const int n = blockIdx.x * 256 + threadIdx.x;   // 0..2047
  const int b = blockIdx.y;
  float acc = 0.f;
  for (int k = 0; k < 256; ++k) acc += Et[b * 256 + k] * Wtd[(size_t)k * 2048 + n];
  for (int k = 0; k < 256; ++k) acc += Ed[b * 256 + k] * Wtd[(size_t)(256 + k) * 2048 + n];
  gate[(size_t)b * 2048 + n] = acc;
}

// ---------------- WMMA GEMM: C[M,N] = A[M,K](bf16) @ W[K,N](bf16) ----------------
// BM=64, BN=128, Kc=32; 256 threads = 8 waves (4 x 2); optional per-(b,col) gate; bf16 or f32 out.
// A tile staged via TDM (each wave DMAs its 8 rows); W tile staged transposed by VALU loads.
template <bool GATED, bool OUT_BF16>
__global__ __launch_bounds__(256) void gemm_wmma(
    const bf16_t* __restrict__ A, const bf16_t* __restrict__ W,
    void* __restrict__ Out, const float* __restrict__ gate,
    int M, int N, int K, int Srows) {
  __shared__ __align__(16) bf16_t sA[64 * 32];
  __shared__ __align__(16) bf16_t sBt[128 * 32];   // transposed [n][k]
  const int tid   = threadIdx.x;
  const int lane  = tid & 31;
  const int wave  = __builtin_amdgcn_readfirstlane(tid >> 5);  // uniform 0..7
  const int wm    = wave >> 1;           // 0..3
  const int wn    = wave & 1;            // 0..1
  const int m0    = blockIdx.y * 64;
  const int n0    = blockIdx.x * 128;
  const int l16   = lane & 15;
  const int khalf = (lane >> 4) * 8;
  const int kb16  = (lane >> 4) * 16;

  v8f acc[4] = {};
  const int nK = K >> 5;
  for (int kc = 0; kc < nK; ++kc) {
    __syncthreads();
#ifdef CDNA5_HAS_TDM
    // each wave DMAs rows [wave*8, wave*8+8) x 32 cols of the A tile into LDS
    tdm_load_2d_bf16((unsigned)(uintptr_t)(&sA[0]) + (unsigned)(wave * 8 * 32 * 2),
                     &A[(size_t)(m0 + wave * 8) * K + (size_t)kc * 32],
                     32u, 8u, (unsigned)K);
#else
    {  // A tile 64x32, 8 bf16 per thread
      int idx = tid * 8;
      int r = idx >> 5, c = idx & 31;
      *(v8bf*)&sA[idx] = *(const v8bf*)&A[(size_t)(m0 + r) * K + (size_t)kc * 32 + c];
    }
#endif
    {  // W tile 32x128 -> sBt[col][k]
      int idx = tid * 16;
      int r = idx >> 7, c = idx & 127;
      const bf16_t* src = &W[(size_t)(kc * 32 + r) * N + n0 + c];
#pragma unroll
      for (int i = 0; i < 16; ++i) sBt[(c + i) * 32 + r] = src[i];
      if (kc + 1 < nK) {  // prefetch next chunk's lines (global_prefetch_b8)
        __builtin_prefetch(&W[(size_t)((kc + 1) * 32 + r) * N + n0 + c], 0, 3);
        __builtin_prefetch(&A[(size_t)(m0 + (tid >> 2)) * K + (size_t)(kc + 1) * 32 + ((tid & 3) * 8)], 0, 3);
      }
    }
#ifdef CDNA5_HAS_TDM
    __builtin_amdgcn_s_wait_tensorcnt(0);
#endif
    __syncthreads();
    const int arow = wm * 16 + l16;
    v16bf af = load_fragA(&sA[arow * 32 + khalf], &sA[arow * 32 + 16 + khalf]);
#pragma unroll
    for (int ct = 0; ct < 4; ++ct) {
      int bcol = wn * 64 + ct * 16 + l16;
      v16bf bfr = load_fragB(&sBt[bcol * 32 + kb16]);
      acc[ct] = wmma_bf16f32(af, bfr, acc[ct]);
    }
  }
  const int rhalf = (lane >> 4) * 8;
#pragma unroll
  for (int ct = 0; ct < 4; ++ct) {
    int col = n0 + wn * 64 + ct * 16 + l16;
#pragma unroll
    for (int r = 0; r < 8; ++r) {
      int row = m0 + wm * 16 + r + rhalf;
      float v = acc[ct][r];
      if (GATED) v *= gate[(size_t)(row / Srows) * N + col];
      if (OUT_BF16) ((bf16_t*)Out)[(size_t)row * N + col] = (bf16_t)v;
      else          ((float*)Out)[(size_t)row * N + col] = v;
    }
  }
}

// ---------------- attention: per (b, h, 32-query tile) ----------------
__global__ __launch_bounds__(128) void attn_wmma(
    const bf16_t* __restrict__ Qg, const bf16_t* __restrict__ Kg,
    const bf16_t* __restrict__ Vg, bf16_t* __restrict__ Ob) {
  constexpr int S = 200, HD = 2048, KVP = 224;       // KVP = 7*32 padded kv
  __shared__ __align__(16) float  sS[32 * KVP];      // 28 KB scores
  __shared__ __align__(16) bf16_t sP[32 * KVP];      // 14 KB probabilities
  __shared__ __align__(16) bf16_t sVt[256 * 40];     // 20 KB V^T chunk, row stride 40
  const int tid   = threadIdx.x;
  const int lane  = tid & 31;
  const int wave  = __builtin_amdgcn_readfirstlane(tid >> 5);  // uniform 0..3
  const int waveq = wave >> 1;       // query strip 0..1 (16 rows each)
  const int wavet = wave & 1;        // kv-tile / d-half split
  const int l16   = lane & 15;
  const int khalf = (lane >> 4) * 8;
  const int kb16  = (lane >> 4) * 16;
  const int q0 = blockIdx.x * 32;
  const int h  = blockIdx.y, b = blockIdx.z;
  const size_t base = ((size_t)b * S) * HD + (size_t)h * 256;

  // Phase 1: scores = (Q K^T) / sqrt(256)
  int qrow = q0 + waveq * 16 + l16;
  if (qrow >= S) qrow = S - 1;                 // clamped; masked at store
  v16bf qa[8];
  {
    const bf16_t* qp = Qg + base + (size_t)qrow * HD;
#pragma unroll
    for (int c = 0; c < 8; ++c)
      qa[c] = load_fragA(qp + c * 32 + khalf, qp + c * 32 + 16 + khalf);
  }
  for (int t = wavet; t < 13; t += 2) {        // scalar loop: t is wave-uniform
    int krow = t * 16 + l16;
    if (krow >= S) krow = S - 1;               // clamped; cols >= S never read
    const bf16_t* kp = Kg + base + (size_t)krow * HD;
    v8f acc = {};
#pragma unroll
    for (int c = 0; c < 8; ++c) {
      v16bf kf = load_fragB(kp + c * 32 + kb16);
      acc = wmma_bf16f32(qa[c], kf, acc);
    }
#pragma unroll
    for (int r = 0; r < 8; ++r) {
      int row = waveq * 16 + r + (lane >> 4) * 8;
      sS[row * KVP + t * 16 + l16] = acc[r] * 0.0625f;
    }
  }
  __syncthreads();

  // Phase 2: per-row softmax, emit zero-padded bf16 P
  if (tid < 32) {
    float* rowp = &sS[tid * KVP];
    float mx = -1e30f;
    for (int j = 0; j < S; ++j) mx = fmaxf(mx, rowp[j]);
    float sum = 0.f;
    for (int j = 0; j < S; ++j) { float e = __expf(rowp[j] - mx); rowp[j] = e; sum += e; }
    float inv = 1.f / sum;
    bf16_t* pp = &sP[tid * KVP];
    for (int j = 0; j < S; ++j) pp[j] = (bf16_t)(rowp[j] * inv);
    for (int j = S; j < KVP; ++j) pp[j] = (bf16_t)0.f;
  }
  __syncthreads();

  // Phase 3: O = P @ V  (each wave: 16 q rows x 128 d cols)
  v8f oacc[8] = {};
  for (int cc = 0; cc < 7; ++cc) {
    if (cc) __syncthreads();
    // stage V rows [cc*32, cc*32+32) transposed into sVt[d][k]
#pragma unroll
    for (int rr = 0; rr < 8; ++rr) {
      int linear = (tid + rr * 128) * 8;       // 0..8191
      int k = linear >> 8;
      int d = linear & 255;
      int kv = cc * 32 + k;
      v8bf vv = {};
      if (kv < S) vv = *(const v8bf*)(Vg + base + (size_t)kv * HD + d);
#pragma unroll
      for (int i = 0; i < 8; ++i) sVt[(d + i) * 40 + k] = vv[i];
    }
    __syncthreads();
    const int prow = waveq * 16 + l16;
    v16bf pa = load_fragA(&sP[prow * KVP + cc * 32 + khalf],
                          &sP[prow * KVP + cc * 32 + 16 + khalf]);
#pragma unroll
    for (int dt = 0; dt < 8; ++dt) {
      int d = wavet * 128 + dt * 16 + l16;
      v16bf vf = load_fragB(&sVt[d * 40 + kb16]);
      oacc[dt] = wmma_bf16f32(pa, vf, oacc[dt]);
    }
  }
  // store O (bf16) in [B,S,H,D] layout
#pragma unroll
  for (int dt = 0; dt < 8; ++dt) {
    int d = wavet * 128 + dt * 16 + l16;
#pragma unroll
    for (int r = 0; r < 8; ++r) {
      int q = q0 + waveq * 16 + r + (lane >> 4) * 8;
      if (q < S) Ob[base + (size_t)q * HD + d] = (bf16_t)oacc[dt][r];
    }
  }
}

extern "C" void kernel_launch(void* const* d_in, const int* in_sizes, int n_in,
                              void* d_out, int out_size, void* d_ws, size_t ws_size,
                              hipStream_t stream) {
  (void)in_sizes; (void)n_in; (void)out_size; (void)ws_size;
  const float* E_h   = (const float*)d_in[0];
  const float* E_t   = (const float*)d_in[1];
  const float* E_d   = (const float*)d_in[2];
  const float* W_Q   = (const float*)d_in[3];
  const float* W_K   = (const float*)d_in[4];
  const float* W_V   = (const float*)d_in[5];
  const float* W_td  = (const float*)d_in[6];
  const float* W_out = (const float*)d_in[7];
  float* out = (float*)d_out;

  constexpr int B = 128, S = 200, D = 256, H = 8;
  constexpr int M = B * S;   // 25600
  constexpr int N = D * H;   // 2048

  char* ws = (char*)d_ws;
  size_t off = 0;
  auto take = [&](size_t bytes) {
    void* p = ws + off;
    off += (bytes + 255) & ~(size_t)255;
    return p;
  };
  bf16_t* Ehb  = (bf16_t*)take((size_t)M * D * 2);
  bf16_t* Wqb  = (bf16_t*)take((size_t)D * N * 2);
  bf16_t* Wkb  = (bf16_t*)take((size_t)D * N * 2);
  bf16_t* Wvb  = (bf16_t*)take((size_t)D * N * 2);
  bf16_t* Wob  = (bf16_t*)take((size_t)N * D * 2);
  float*  gate = (float*) take((size_t)B * N * 4);
  bf16_t* Qg   = (bf16_t*)take((size_t)M * N * 2);
  bf16_t* Kg   = (bf16_t*)take((size_t)M * N * 2);
  bf16_t* Vg   = (bf16_t*)take((size_t)M * N * 2);
  bf16_t* At   = (bf16_t*)take((size_t)M * N * 2);

  auto cvt = [&](const float* s, bf16_t* d, int n) {
    cvt_bf16<<<dim3((n + 1023) / 1024), 256, 0, stream>>>(s, d, n);
  };
  cvt(E_h,   Ehb, M * D);
  cvt(W_Q,   Wqb, D * N);
  cvt(W_K,   Wkb, D * N);
  cvt(W_V,   Wvb, D * N);
  cvt(W_out, Wob, N * D);

  gate_kernel<<<dim3(N / 256, B), 256, 0, stream>>>(E_t, E_d, W_td, gate);

  gemm_wmma<true, true><<<dim3(N / 128, M / 64), 256, 0, stream>>>(Ehb, Wqb, Qg, gate, M, N, D, S);
  gemm_wmma<true, true><<<dim3(N / 128, M / 64), 256, 0, stream>>>(Ehb, Wkb, Kg, gate, M, N, D, S);
  gemm_wmma<true, true><<<dim3(N / 128, M / 64), 256, 0, stream>>>(Ehb, Wvb, Vg, gate, M, N, D, S);

  attn_wmma<<<dim3((S + 31) / 32, H, B), 128, 0, stream>>>(Qg, Kg, Vg, At);

  gemm_wmma<false, false><<<dim3(D / 128, M / 64), 256, 0, stream>>>(At, Wob, out, nullptr, M, D, N, S);
}
